// VRNN_41755672052459
// MI455X (gfx1250) — matrix operand order repaired and checked by reading, not compiled, not run on hardware
//
#include <hip/hip_runtime.h>
#include <hip/hip_bf16.h>

// VRNN forward for MI455X (gfx1250, wave32).
// Recurrence is independent per batch row -> 16 persistent workgroups, each
// owning a 16-row batch tile across all 512 timesteps. All GEMMs use
// v_wmma_f32_16x16x32_bf16. Activations live in LDS; weights are converted
// once per launch to bf16 in d_ws (~2.3MB, L2-resident at 192MB) and streamed
// as B fragments. Independent GEMMs are co-scheduled in the same barrier
// phase to keep all 32 waves busy (the sequential time loop makes per-step
// latency the bottleneck, not bandwidth: outputs ~200MB vs 23.3TB/s HBM).

#define SEQL 512
#define BATCH 256
#define OBSD 128
#define LATD 64
#define HIDD 256

typedef __bf16 bf16;
typedef __attribute__((ext_vector_type(16))) bf16 v16bf;
typedef __attribute__((ext_vector_type(4))) bf16 v4bf;
typedef __attribute__((ext_vector_type(8))) float v8f;

__device__ __forceinline__ float sigmoidf(float x) {
  return 1.0f / (1.0f + __expf(-x));
}

struct Params {
  const float* x;
  const float* eps;
  const bf16* w[11];   // bf16 weights in workspace, [N][K] row-major
  const float* b[11];  // fp32 biases (device inputs)
  float* out;
};

// ---- WMMA fragment loaders -------------------------------------------------
// A: 16xK row-major bf16 in LDS. ISA layout (16-bit A 16x32):
//   lanes 0-15 : M=lane,    elems 0-7 = K 0-7,  elems 8-15 = K 16-23
//   lanes 16-31: M=lane-16, elems 0-7 = K 8-15, elems 8-15 = K 24-31
__device__ __forceinline__ v16bf load_A(const bf16* aL, int lda, int kBase,
                                        int lane) {
  const int m = lane & 15;
  const int kh = ((lane >> 4) & 1) * 8;
  const bf16* r = aL + m * lda + kBase + kh;
  v16bf a;
  ((uint4*)&a)[0] = *(const uint4*)(r);       // K = kh .. kh+7
  ((uint4*)&a)[1] = *(const uint4*)(r + 16);  // K = 16+kh .. 16+kh+7
  return a;
}

// B: weight [N][K] row-major bf16 in global (L2 resident).
// ISA layout (16-bit B 32x16): lanes 0-15 hold K 0-7,16-23 (N=lane),
// lanes 16-31 hold K 8-15,24-31 (N=lane-16); two 16B loads per lane.
__device__ __forceinline__ v16bf load_B(const bf16* w, int Kw, int nBase,
                                        int kBase, int lane) {
  const int n = nBase + (lane & 15);
  const bf16* r = w + (size_t)n * Kw + kBase + ((lane >> 4) << 3);
  v16bf b;
  ((uint4*)&b)[0] = *(const uint4*)(r);
  ((uint4*)&b)[1] = *(const uint4*)(r + 16);
  return b;
}

__device__ __forceinline__ v8f bias_acc(const float* bias, int nBase, int lane) {
  const float bv = bias[nBase + (lane & 15)];
  v8f acc;
#pragma unroll
  for (int r = 0; r < 8; ++r) acc[r] = bv;
  return acc;
}

// acc += A(16xKlen from LDS) * B(Klen x 16 slice of weight row length Kw,
// starting at K-offset kw0 in the weight).
__device__ __forceinline__ v8f gemm_acc(v8f acc, const bf16* aL, int lda,
                                        int Klen, const bf16* wG, int Kw,
                                        int kw0, int nBase, int lane) {
#pragma unroll 4
  for (int k = 0; k < Klen; k += 32) {
    v16bf a = load_A(aL, lda, k, lane);
    v16bf b = load_B(wG, Kw, nBase, kw0 + k, lane);
    acc = __builtin_amdgcn_wmma_f32_16x16x32_bf16(false, a, false, b,
                                                  (short)0, acc, false, false);
  }
  return acc;
}

// C layout: N = lane&15, VGPR r holds M = r + 8*(lane>>4).
__device__ __forceinline__ void store_bf16_lds(v8f acc, bf16* dst, int ldn,
                                               int nBase, int lane, bool relu) {
  const int n = nBase + (lane & 15);
  const int mh = (lane >> 4) * 8;
#pragma unroll
  for (int r = 0; r < 8; ++r) {
    float v = acc[r];
    if (relu) v = fmaxf(v, 0.f);          // v_max_num_f32
    dst[(mh + r) * ldn + n] = (bf16)v;    // native f32->bf16 cvt
  }
}

__device__ __forceinline__ void store_f32_lds(v8f acc, float* dst, int ldn,
                                              int nBase, int lane) {
  const int n = nBase + (lane & 15);
  const int mh = (lane >> 4) * 8;
#pragma unroll
  for (int r = 0; r < 8; ++r) dst[(mh + r) * ldn + n] = acc[r];
}

// ---- weight fp32 -> bf16 conversion ---------------------------------------
__global__ void cvt_bf16(const float* __restrict__ src, bf16* __restrict__ dst,
                         int n) {
  int i = blockIdx.x * blockDim.x + threadIdx.x;
  if (i < n) dst[i] = (bf16)src[i];
}

// ---- main persistent VRNN kernel ------------------------------------------
__global__ __launch_bounds__(1024, 1) void vrnn_kernel(Params p) {
  __shared__ __align__(16) bf16 sXbf[16 * OBSD];   // x_t tile
  __shared__ __align__(16) bf16 sHbf[16 * HIDD];   // h (bf16)
  __shared__ __align__(16) bf16 sT0[16 * HIDD];    // phi_x / enc / dec chain
  __shared__ __align__(16) bf16 sT1[16 * HIDD];    // prior chain
  __shared__ __align__(16) bf16 sPhiX[16 * HIDD];
  __shared__ __align__(16) bf16 sPhiZ[16 * HIDD];
  __shared__ __align__(16) bf16 sZ[16 * LATD];
  __shared__ __align__(16) float sHf[16 * HIDD];   // h (f32)
  __shared__ __align__(16) float sPost[16 * (2 * LATD)];
  __shared__ __align__(16) float sRZ[16 * (2 * HIDD)];  // xr+hr | xz+hz
  __shared__ __align__(16) float sXN[16 * HIDD];
  __shared__ __align__(16) float sHN[16 * HIDD];

  const int tid = threadIdx.x;
  const int lane = tid & 31;
  const int wv = tid >> 5;
  const int NW = blockDim.x >> 5;  // 32 waves
  const int b0 = blockIdx.x * 16;  // batch tile base

  const size_t OFF_REC = 0;
  const size_t OFF_PRM = (size_t)SEQL * BATCH * OBSD;
  const size_t OFF_PRLV = OFF_PRM + (size_t)SEQL * BATCH * LATD;
  const size_t OFF_POM = OFF_PRLV + (size_t)SEQL * BATCH * LATD;
  const size_t OFF_POLV = OFF_POM + (size_t)SEQL * BATCH * LATD;

  // h0 = 0
  for (int i = tid; i < 16 * HIDD; i += blockDim.x) {
    sHf[i] = 0.f;
    sHbf[i] = (bf16)0.f;
  }
  __syncthreads();

  for (int t = 0; t < SEQL; ++t) {
    // --- P0: stage x_t tile (float4 -> 4x bf16); prefetch next step ---
    if (tid < 512) {
      int m = tid >> 5, n4 = (tid & 31) * 4;
      float4 f = *(const float4*)(p.x + ((size_t)t * BATCH + b0 + m) * OBSD + n4);
      v4bf v = {(bf16)f.x, (bf16)f.y, (bf16)f.z, (bf16)f.w};
      *(v4bf*)(sXbf + m * OBSD + n4) = v;
    }
    if (t + 1 < SEQL && tid < 128)
      __builtin_prefetch(p.x + ((size_t)(t + 1) * BATCH + b0) * OBSD + tid * 16,
                         0, 1);
    __syncthreads();

    // --- P1 (32 tiles): t0x = relu(x@W_phi1^T+b) || t1h = relu(h@W_pr1^T+b)
    for (int nt = wv; nt < 32; nt += NW) {
      if (nt < 16) {
        int nB = nt * 16;
        v8f acc = bias_acc(p.b[0], nB, lane);
        acc = gemm_acc(acc, sXbf, OBSD, OBSD, p.w[0], OBSD, 0, nB, lane);
        store_bf16_lds(acc, sT0, HIDD, nB, lane, true);
      } else {
        int nB = (nt - 16) * 16;
        v8f acc = bias_acc(p.b[3], nB, lane);
        acc = gemm_acc(acc, sHbf, HIDD, HIDD, p.w[3], HIDD, 0, nB, lane);
        store_bf16_lds(acc, sT1, HIDD, nB, lane, true);
      }
    }
    __syncthreads();

    // --- P2 (24 tiles): phi_x = t0x@W_phi2^T+b || prior = t1h@W_pr2^T+b ---
    for (int nt = wv; nt < 24; nt += NW) {
      if (nt < 16) {
        int nB = nt * 16;
        v8f acc = bias_acc(p.b[1], nB, lane);
        acc = gemm_acc(acc, sT0, HIDD, HIDD, p.w[1], HIDD, 0, nB, lane);
        store_bf16_lds(acc, sPhiX, HIDD, nB, lane, false);
      } else {
        int nB = (nt - 16) * 16;
        v8f acc = bias_acc(p.b[4], nB, lane);
        acc = gemm_acc(acc, sT1, HIDD, HIDD, p.w[4], HIDD, 0, nB, lane);
        int n = nB + (lane & 15);
        int mh = (lane >> 4) * 8;
        float* dst = p.out + (n < LATD ? OFF_PRM : OFF_PRLV);
        int nn = n < LATD ? n : n - LATD;
#pragma unroll
        for (int r = 0; r < 8; ++r)
          dst[((size_t)t * BATCH + b0 + mh + r) * LATD + nn] = acc[r];
      }
    }
    __syncthreads();

    // --- P3 (16 tiles): enc1 = relu([phi_x|h]@W_e1^T+b)  (K=512) ---
    for (int nt = wv; nt < 16; nt += NW) {
      int nB = nt * 16;
      v8f acc = bias_acc(p.b[5], nB, lane);
      acc = gemm_acc(acc, sPhiX, HIDD, HIDD, p.w[5], 2 * HIDD, 0, nB, lane);
      acc = gemm_acc(acc, sHbf, HIDD, HIDD, p.w[5], 2 * HIDD, HIDD, nB, lane);
      store_bf16_lds(acc, sT0, HIDD, nB, lane, true);
    }
    __syncthreads();

    // --- P4 (56 tiles): post = enc1@W_e2^T+b || gh = h@W_hh^T+bhh (N=768) ---
    for (int nt = wv; nt < 56; nt += NW) {
      if (nt < 8) {
        int nB = nt * 16;
        v8f acc = bias_acc(p.b[6], nB, lane);
        acc = gemm_acc(acc, sT0, HIDD, HIDD, p.w[6], HIDD, 0, nB, lane);
        store_f32_lds(acc, sPost, 2 * LATD, nB, lane);
      } else {
        int nB = (nt - 8) * 16;
        v8f ah = bias_acc(p.b[10], nB, lane);
        ah = gemm_acc(ah, sHbf, HIDD, HIDD, p.w[10], HIDD, 0, nB, lane);
        int n = nB + (lane & 15);
        int mh = (lane >> 4) * 8;
        if (n < 2 * HIDD) {  // hr | hz
#pragma unroll
          for (int r = 0; r < 8; ++r) sRZ[(mh + r) * (2 * HIDD) + n] = ah[r];
        } else {             // hn kept separate (r * hn later)
          int nn = n - 2 * HIDD;
#pragma unroll
          for (int r = 0; r < 8; ++r) sHN[(mh + r) * HIDD + nn] = ah[r];
        }
      }
    }
    __syncthreads();

    // --- P5: z = mean + eps*exp(0.5*lv); emit po_m / po_lv (1024 elems) ---
    {
      int m = tid >> 6, n = tid & 63;
      float mean = sPost[m * (2 * LATD) + n];
      float lv = sPost[m * (2 * LATD) + LATD + n];
      size_t gi = ((size_t)t * BATCH + b0 + m) * LATD + n;
      p.out[OFF_POM + gi] = mean;
      p.out[OFF_POLV + gi] = lv;
      float z = mean + p.eps[gi] * __expf(0.5f * lv);
      sZ[m * LATD + n] = (bf16)z;
    }
    __syncthreads();

    // --- P6 (16 tiles): phi_z = relu(z@W_phz^T+b)  (K=64) ---
    for (int nt = wv; nt < 16; nt += NW) {
      int nB = nt * 16;
      v8f acc = bias_acc(p.b[2], nB, lane);
      acc = gemm_acc(acc, sZ, LATD, LATD, p.w[2], LATD, 0, nB, lane);
      store_bf16_lds(acc, sPhiZ, HIDD, nB, lane, true);
    }
    __syncthreads();

    // --- P7 (64 tiles): dec1 = relu([phi_z|h]@W_d1^T+b) ||
    //                    gx = [phi_x|phi_z]@W_ih^T+bih (accumulate into gates)
    for (int nt = wv; nt < 64; nt += NW) {
      if (nt < 16) {
        int nB = nt * 16;
        v8f acc = bias_acc(p.b[7], nB, lane);
        acc = gemm_acc(acc, sPhiZ, HIDD, HIDD, p.w[7], 2 * HIDD, 0, nB, lane);
        acc = gemm_acc(acc, sHbf, HIDD, HIDD, p.w[7], 2 * HIDD, HIDD, nB, lane);
        store_bf16_lds(acc, sT0, HIDD, nB, lane, true);
      } else {
        int nB = (nt - 16) * 16;
        v8f ax = bias_acc(p.b[9], nB, lane);
        ax = gemm_acc(ax, sPhiX, HIDD, HIDD, p.w[9], 2 * HIDD, 0, nB, lane);
        ax = gemm_acc(ax, sPhiZ, HIDD, HIDD, p.w[9], 2 * HIDD, HIDD, nB, lane);
        int n = nB + (lane & 15);
        int mh = (lane >> 4) * 8;
        if (n < 2 * HIDD) {  // xr+hr | xz+hz
#pragma unroll
          for (int r = 0; r < 8; ++r) sRZ[(mh + r) * (2 * HIDD) + n] += ax[r];
        } else {
          int nn = n - 2 * HIDD;
#pragma unroll
          for (int r = 0; r < 8; ++r) sXN[(mh + r) * HIDD + nn] = ax[r];
        }
      }
    }
    __syncthreads();

    // --- P8: recon = dec1@W_d2^T+b (waves 0-7) || GRU h update (rest) ---
    if (wv < 8) {
      int nB = wv * 16;
      v8f acc = bias_acc(p.b[8], nB, lane);
      acc = gemm_acc(acc, sT0, HIDD, HIDD, p.w[8], HIDD, 0, nB, lane);
      int n = nB + (lane & 15);
      int mh = (lane >> 4) * 8;
#pragma unroll
      for (int r = 0; r < 8; ++r)
        p.out[OFF_REC + ((size_t)t * BATCH + b0 + mh + r) * OBSD + n] = acc[r];
    } else {
      for (int i = tid - 256; i < 16 * HIDD; i += 768) {
        int m = i >> 8, n = i & 255;
        float rr = sigmoidf(sRZ[m * (2 * HIDD) + n]);
        float zg = sigmoidf(sRZ[m * (2 * HIDD) + HIDD + n]);
        float nn = tanhf(sXN[i] + rr * sHN[i]);
        float hv = (1.f - zg) * nn + zg * sHf[i];
        sHf[i] = hv;
        sHbf[i] = (bf16)hv;
      }
    }
    __syncthreads();
  }
}

// ---------------------------------------------------------------------------
extern "C" void kernel_launch(void* const* d_in, const int* in_sizes, int n_in,
                              void* d_out, int out_size, void* d_ws,
                              size_t ws_size, hipStream_t stream) {
  (void)n_in; (void)out_size; (void)ws_size;
  // setup_inputs order:
  // 0 x, 1 eps, 2 phi_x_w1, 3 phi_x_b1, 4 phi_x_w2, 5 phi_x_b2, 6 phi_z_w,
  // 7 phi_z_b, 8 prior_w1, 9 prior_b1, 10 prior_w2, 11 prior_b2, 12 enc_w1,
  // 13 enc_b1, 14 enc_w2, 15 enc_b2, 16 dec_w1, 17 dec_b1, 18 dec_w2,
  // 19 dec_b2, 20 gru_wih, 21 gru_whh, 22 gru_bih, 23 gru_bhh
  static const int widx[11] = {2, 4, 6, 8, 10, 12, 14, 16, 18, 20, 21};
  static const int bidx[11] = {3, 5, 7, 9, 11, 13, 15, 17, 19, 22, 23};

  Params P;
  P.x = (const float*)d_in[0];
  P.eps = (const float*)d_in[1];
  P.out = (float*)d_out;

  bf16* wsp = (bf16*)d_ws;
  size_t off = 0;
  for (int i = 0; i < 11; ++i) {
    const float* src = (const float*)d_in[widx[i]];
    int n = in_sizes[widx[i]];
    cvt_bf16<<<(n + 255) / 256, 256, 0, stream>>>(src, wsp + off, n);
    P.w[i] = wsp + off;
    off += (size_t)n;
    P.b[i] = (const float*)d_in[bidx[i]];
  }

  vrnn_kernel<<<BATCH / 16, 1024, 0, stream>>>(P);
}
